// TrajNet_35338990911698
// MI455X (gfx1250) — matrix operand order repaired and checked by reading, not compiled, not run on hardware
//
#include <hip/hip_runtime.h>
#include <hip/hip_bf16.h>

typedef __attribute__((ext_vector_type(16))) _Float16 v16h;
typedef __attribute__((ext_vector_type(8)))  float    v8f;

#define NB     262144
#define TOBS   20
#define TPRED  30
#define ROWPAD 17   // 16 + 1 pad to reduce LDS bank conflicts

struct WaveLds {
  float h[16][ROWPAD];       // h state staging: [batch row][hid dim]
  float ob[16][TPRED * 2];   // decoder outputs: [batch row][t*2+o]
  float inp[16][2];          // decoder feedback input
};

// Wave-local LDS ordering: LDS ops are in-order per wave (lockstep wave32),
// so a compiler memory fence + dscnt drain is sufficient — no block barrier.
__device__ __forceinline__ void lds_fence() {
  asm volatile("s_wait_dscnt 0" ::: "memory");
}

__device__ __forceinline__ float tanh_f(float x) {
#if __has_builtin(__builtin_amdgcn_tanhf)
  return __builtin_amdgcn_tanhf(x);
#else
  float e = __expf(2.f * x);
  return 1.f - 2.f * __builtin_amdgcn_rcpf(e + 1.f);
#endif
}
// sigmoid(x) = 0.5*tanh(x/2) + 0.5 : single v_tanh_f32 + mul + fma (no rcp/exp)
__device__ __forceinline__ float sigm(float x) {
  return fmaf(0.5f, tanh_f(0.5f * x), 0.5f);
}

// B fragment, 32(K)x16(N) f16. Layout (this code's contiguous K order, same as packA;
// 8 VGPRs): lanes 0-15 hold N=lane, K=0..15 in halves 0..15;
// lanes 16-31 hold N=lane-16, K=16..31.
// Column n of gᵀ uses row n of W' = [wih(2) | whh(16) | 0-pad(14)].
__device__ __forceinline__ v16h buildB(const float* __restrict__ wih,
                                       const float* __restrict__ whh,
                                       int n, bool hi) {
  v16h b;
  if (!hi) {
    b[0] = (_Float16)wih[n * 2 + 0];
    b[1] = (_Float16)wih[n * 2 + 1];
#pragma unroll
    for (int k = 0; k < 14; ++k) b[2 + k] = (_Float16)whh[n * 16 + k];
  } else {
    b[0] = (_Float16)whh[n * 16 + 14];
    b[1] = (_Float16)whh[n * 16 + 15];
#pragma unroll
    for (int k = 2; k < 16; ++k) b[k] = (_Float16)0.f;
  }
  return b;
}

// A fragment, 16(M)x32(K) f16: lanes 0-15: M=lane,
// halves 0..15 = K0..15; lanes 16-31: M=lane-16,
// halves 0..15 = K16..31.
// K slots: 0,1 = emb; 2..17 = h[0..15]; 18..31 = 0.
__device__ __forceinline__ v16h packA(const WaveLds* __restrict__ L, int row,
                                      bool hi, float e0, float e1) {
  v16h a;
  if (!hi) {
    a[0] = (_Float16)e0;
    a[1] = (_Float16)e1;
#pragma unroll
    for (int k = 0; k < 14; ++k) a[2 + k] = (_Float16)L->h[row][k];  // K2..15
  } else {
    a[0] = (_Float16)L->h[row][14];                                 // K16
    a[1] = (_Float16)L->h[row][15];                                 // K17
#pragma unroll
    for (int k = 2; k < 16; ++k) a[k] = (_Float16)0.f;
  }
  return a;
}

__device__ __forceinline__ void stageH(WaveLds* __restrict__ L, const v8f& h,
                                       int nloc, bool hi) {
#pragma unroll
  for (int v = 0; v < 8; ++v) L->h[v + (hi ? 8 : 0)][nloc] = h[v];
}

// One LSTM cell: 4 WMMAs (i,f,g,o gate tiles) + elementwise update.
// h/c live in the 16x16 f32 C/D accumulator layout, so update is per-lane.
__device__ __forceinline__ void cell(const v16h& A, const v16h& B0,
                                     const v16h& B1, const v16h& B2,
                                     const v16h& B3, float cb0, float cb1,
                                     float cb2, float cb3, v8f& h, v8f& c) {
  v8f Ci, Cf, Cg, Co;
#pragma unroll
  for (int j = 0; j < 8; ++j) { Ci[j] = cb0; Cf[j] = cb1; Cg[j] = cb2; Co[j] = cb3; }
  v8f gi = __builtin_amdgcn_wmma_f32_16x16x32_f16(false, A, false, B0, (short)0, Ci, false, false);
  v8f gf = __builtin_amdgcn_wmma_f32_16x16x32_f16(false, A, false, B1, (short)0, Cf, false, false);
  v8f gg = __builtin_amdgcn_wmma_f32_16x16x32_f16(false, A, false, B2, (short)0, Cg, false, false);
  v8f go = __builtin_amdgcn_wmma_f32_16x16x32_f16(false, A, false, B3, (short)0, Co, false, false);
#pragma unroll
  for (int j = 0; j < 8; ++j) {
    float cn = sigm(gf[j]) * c[j] + sigm(gi[j]) * tanh_f(gg[j]);
    c[j] = cn;
    h[j] = sigm(go[j]) * tanh_f(cn);
  }
}

__global__ __launch_bounds__(256) void trajnet_kernel(
    const float* __restrict__ x,
    const float* __restrict__ l1w, const float* __restrict__ l1b,
    const float* __restrict__ l2w, const float* __restrict__ l2b,
    const float* __restrict__ l3w, const float* __restrict__ l3b,
    const float* __restrict__ w1ih, const float* __restrict__ w1hh,
    const float* __restrict__ b1ih, const float* __restrict__ b1hh,
    const float* __restrict__ w2ih, const float* __restrict__ w2hh,
    const float* __restrict__ b2ih, const float* __restrict__ b2hh,
    float* __restrict__ out) {
  __shared__ WaveLds lds[8];
  const int lane = threadIdx.x & 31;
  const int wib  = threadIdx.x >> 5;
  WaveLds* L = &lds[wib];
  const int rowBase = (blockIdx.x * 8 + wib) * 16;
  const int nloc = lane & 15;
  const bool hi = lane >= 16;

  // Tiny dense layers: keep weights in SGPR/VGPR.
  const float w1a = l1w[0], w1b_ = l1w[1], w1c = l1w[2], w1d = l1w[3];
  const float b1a = l1b[0], b1b_ = l1b[1];
  const float w2a = l2w[0], w2b_ = l2w[1], w2c = l2w[2], w2d = l2w[3];
  const float b2a = l2b[0], b2b_ = l2b[1];
  float w3[16], b3;
  {
    const int o = hi ? 1 : 0;
#pragma unroll
    for (int k = 0; k < 16; ++k) w3[k] = l3w[o * 16 + k];
    b3 = l3b[o];
  }

  v8f h = {0, 0, 0, 0, 0, 0, 0, 0};
  v8f c = {0, 0, 0, 0, 0, 0, 0, 0};
  stageH(L, h, nloc, hi);
  lds_fence();

  // ---- encoder: LSTM1 weights -> B fragments (resident in VGPRs) ----
  v16h B0 = buildB(w1ih, w1hh, 0 * 16 + nloc, hi);
  v16h B1 = buildB(w1ih, w1hh, 1 * 16 + nloc, hi);
  v16h B2 = buildB(w1ih, w1hh, 2 * 16 + nloc, hi);
  v16h B3 = buildB(w1ih, w1hh, 3 * 16 + nloc, hi);
  float cb0 = b1ih[0 * 16 + nloc] + b1hh[0 * 16 + nloc];
  float cb1 = b1ih[1 * 16 + nloc] + b1hh[1 * 16 + nloc];
  float cb2 = b1ih[2 * 16 + nloc] + b1hh[2 * 16 + nloc];
  float cb3 = b1ih[3 * 16 + nloc] + b1hh[3 * 16 + nloc];

  // Branchless: lane pair (r, r+16) both compute row r's emb (same address ->
  // broadcast load); only lanes 0-15 actually place it into the A fragment.
  const float* xr = x + (size_t)(rowBase + nloc) * (TOBS * 2);

  for (int t = 0; t < TOBS; ++t) {
    float x0 = xr[t * 2 + 0], x1 = xr[t * 2 + 1];
    float e0 = fmaxf(0.f, x0 * w1a + x1 * w1b_ + b1a);
    float e1 = fmaxf(0.f, x0 * w1c + x1 * w1d + b1b_);
    v16h A = packA(L, nloc, hi, e0, e1);
    cell(A, B0, B1, B2, B3, cb0, cb1, cb2, cb3, h, c);
    lds_fence();               // finish A-pack reads before restage
    stageH(L, h, nloc, hi);
    lds_fence();
  }

  // ---- decoder: LSTM2 weights; dec_in0 = x[:, TOBS-1, :2] ----
  B0 = buildB(w2ih, w2hh, 0 * 16 + nloc, hi);
  B1 = buildB(w2ih, w2hh, 1 * 16 + nloc, hi);
  B2 = buildB(w2ih, w2hh, 2 * 16 + nloc, hi);
  B3 = buildB(w2ih, w2hh, 3 * 16 + nloc, hi);
  cb0 = b2ih[0 * 16 + nloc] + b2hh[0 * 16 + nloc];
  cb1 = b2ih[1 * 16 + nloc] + b2hh[1 * 16 + nloc];
  cb2 = b2ih[2 * 16 + nloc] + b2hh[2 * 16 + nloc];
  cb3 = b2ih[3 * 16 + nloc] + b2hh[3 * 16 + nloc];
  {
    const int o = hi ? 1 : 0;
    L->inp[nloc][o] = xr[(TOBS - 1) * 2 + o];
  }
  lds_fence();

  for (int t = 0; t < TPRED; ++t) {
    float i0 = L->inp[nloc][0], i1 = L->inp[nloc][1];  // broadcast reads
    float e0 = fmaxf(0.f, i0 * w2a + i1 * w2b_ + b2a);
    float e1 = fmaxf(0.f, i0 * w2c + i1 * w2d + b2b_);
    v16h A = packA(L, nloc, hi, e0, e1);
    cell(A, B0, B1, B2, B3, cb0, cb1, cb2, cb3, h, c);
    lds_fence();
    stageH(L, h, nloc, hi);    // h_out now in LDS (also serves next A-pack)
    lds_fence();
    {  // out = h_out @ l3wᵀ + b ; lane pair (r, r+16) computes (out0, out1)
      float acc = b3;
#pragma unroll
      for (int k = 0; k < 16; ++k) acc += L->h[nloc][k] * w3[k];
      const int o = hi ? 1 : 0;
      L->ob[nloc][t * 2 + o] = acc;
      L->inp[nloc][o] = acc;
    }
    lds_fence();
  }

  // Coalesced flush: this wave owns 16 rows * 60 floats contiguous in out.
  float* ob = out + (size_t)rowBase * (TPRED * 2);
  const float* src = &L->ob[0][0];
  for (int i = lane; i < 16 * TPRED * 2; i += 32) ob[i] = src[i];
}

extern "C" void kernel_launch(void* const* d_in, const int* in_sizes, int n_in,
                              void* d_out, int out_size, void* d_ws, size_t ws_size,
                              hipStream_t stream) {
  (void)in_sizes; (void)n_in; (void)out_size; (void)d_ws; (void)ws_size;
  const float* x    = (const float*)d_in[0];
  // d_in[1] fixed_params, d_in[2] var_inp: unused by the reference
  const float* l1w  = (const float*)d_in[3];
  const float* l1b  = (const float*)d_in[4];
  const float* l2w  = (const float*)d_in[5];
  const float* l2b  = (const float*)d_in[6];
  const float* l3w  = (const float*)d_in[7];
  const float* l3b  = (const float*)d_in[8];
  const float* w1ih = (const float*)d_in[9];
  const float* w1hh = (const float*)d_in[10];
  const float* b1ih = (const float*)d_in[11];
  const float* b1hh = (const float*)d_in[12];
  const float* w2ih = (const float*)d_in[13];
  const float* w2hh = (const float*)d_in[14];
  const float* b2ih = (const float*)d_in[15];
  const float* b2hh = (const float*)d_in[16];
  float* out = (float*)d_out;

  dim3 grid(NB / (16 * 8));  // 16 rows/wave, 8 waves/block -> 2048 blocks
  trajnet_kernel<<<grid, 256, 0, stream>>>(x, l1w, l1b, l2w, l2b, l3w, l3b,
                                           w1ih, w1hh, b1ih, b1hh,
                                           w2ih, w2hh, b2ih, b2hh, out);
}
